// da_retinanet_68779606278586
// MI455X (gfx1250) — matrix operand order repaired and checked by reading, not compiled, not run on hardware
//
#include <hip/hip_runtime.h>
#include <hip/hip_bf16.h>

typedef __attribute__((ext_vector_type(16))) _Float16 v16h;
typedef __attribute__((ext_vector_type(8)))  float    v8f;

#if defined(__has_builtin)
#if __has_builtin(__builtin_amdgcn_tensor_load_to_lds) && __has_builtin(__builtin_amdgcn_s_wait_tensorcnt)
#define DA_HAS_TDM 1
#endif
#endif
#ifndef DA_HAS_TDM
#define DA_HAS_TDM 0
#endif

// ---------------------------------------------------------------------------
// helpers
// ---------------------------------------------------------------------------
#define PE_COEF (9.210340371976184f / 256.0f)   // ln(10000)/D

__device__ __forceinline__ float pe_val(int l, int d) {
    int i = d >> 1;
    float div = __expf(-(float)(2 * i) * PE_COEF);
    float ang = (float)l * div;
    return (d & 1) ? __cosf(ang) : __sinf(ang);
}

// convert four float4s to the 16 packed halves of a WMMA fragment
__device__ __forceinline__ void cvt16(v16h& dst, float4 x0, float4 x1, float4 x2, float4 x3) {
    dst[0]=(_Float16)x0.x;  dst[1]=(_Float16)x0.y;  dst[2]=(_Float16)x0.z;  dst[3]=(_Float16)x0.w;
    dst[4]=(_Float16)x1.x;  dst[5]=(_Float16)x1.y;  dst[6]=(_Float16)x1.z;  dst[7]=(_Float16)x1.w;
    dst[8]=(_Float16)x2.x;  dst[9]=(_Float16)x2.y;  dst[10]=(_Float16)x2.z; dst[11]=(_Float16)x2.w;
    dst[12]=(_Float16)x3.x; dst[13]=(_Float16)x3.y; dst[14]=(_Float16)x3.z; dst[15]=(_Float16)x3.w;
}
__device__ __forceinline__ void unpack4(float* d, float4 v) { d[0]=v.x; d[1]=v.y; d[2]=v.z; d[3]=v.w; }

// ---------------------------------------------------------------------------
// PE table generator: pe[l*256+d] (optional) and peT[d*L+l]
// ---------------------------------------------------------------------------
__global__ void da_pe_gen(float* __restrict__ pe, float* __restrict__ peT, int L)
{
    long idx = (long)blockIdx.x * 256 + threadIdx.x;
    if (idx >= (long)L * 256) return;
    int l = (int)(idx >> 8);
    int d = (int)(idx & 255);
    float v = pe_val(l, d);
    if (pe) pe[idx] = v;
    peT[(long)d * L + l] = v;
}

// ---------------------------------------------------------------------------
// Generic strided NT GEMM: out[m,n] = sum_k A[m,k]*B[n,k] (+bias) (+PE table on A)
// K must be a multiple of 32. One wave -> one 16x16 tile; 4 waves / block.
// AFAST/BFAST: compile-time "K-stride == 1, 16B-aligned rows" vector paths.
// ---------------------------------------------------------------------------
template<int AFAST, int BFAST, int PEA>
__global__ void da_gemm_nt_wmma(const float* __restrict__ A, long aBatch, long aM, long aK,
                                const float* __restrict__ Bm, long bBatch, long bN, long bK,
                                const float* __restrict__ bias, int biasMode, // 0 none,1 per-m,2 per-n
                                const float* __restrict__ peTab,             // pe[m*256+k]
                                float* __restrict__ Out, long oBatch, long oM,
                                int M, int N, int K)
{
    int lane = threadIdx.x & 31;
    int wid  = threadIdx.x >> 5;
    int half = lane >> 4;
    int nrow = lane & 15;
    int tn_count = (N + 15) >> 4;
    int tm_count = (M + 15) >> 4;
    int tile = blockIdx.x * 4 + wid;
    if (tile >= tm_count * tn_count) return;
    int tm = tile / tn_count;
    int tn = tile - tm * tn_count;

    const float* Ab = A  + (long)blockIdx.y * aBatch;
    const float* Bb = Bm + (long)blockIdx.y * bBatch;
    float*       Ob = Out + (long)blockIdx.y * oBatch;

    int m = tm * 16 + nrow;  int mc = (m < M) ? m : (M - 1);
    int n = tn * 16 + nrow;  int nc = (n < N) ? n : (N - 1);
    const float* Ap = Ab + (long)mc * aM;       // clamped row: loads never fault,
    const float* Bp = Bb + (long)nc * bN;       // garbage stays in unstored rows/cols
    const float* Av = Ap + (long)(half * 8)  * aK;
    const float* Bv = Bp + (long)(half * 8) * bK;

    v8f c;
#pragma unroll
    for (int j = 0; j < 8; ++j) c[j] = 0.0f;

    for (int kc = 0; kc < K; kc += 32) {
        float av[16], bv[16];
        if (AFAST) {
            const float4* p = (const float4*)(Ap + kc + half * 8);
            __builtin_prefetch(p + 16, 0, 1);
            unpack4(av + 0, p[0]); unpack4(av + 4, p[1]);
            unpack4(av + 8, p[4]); unpack4(av + 12, p[5]);
        } else {
#pragma unroll
            for (int i = 0; i < 16; ++i) {
                int di = (i < 8) ? i : i + 8;
                av[i] = Av[(long)(kc + di) * aK];
            }
        }
        if (PEA) {
            const float4* pp = (const float4*)(peTab + (long)mc * 256 + kc + half * 8);
            float4 p0 = pp[0], p1 = pp[1], p2 = pp[4], p3 = pp[5];
            av[0]+=p0.x; av[1]+=p0.y; av[2]+=p0.z; av[3]+=p0.w;
            av[4]+=p1.x; av[5]+=p1.y; av[6]+=p1.z; av[7]+=p1.w;
            av[8]+=p2.x; av[9]+=p2.y; av[10]+=p2.z; av[11]+=p2.w;
            av[12]+=p3.x; av[13]+=p3.y; av[14]+=p3.z; av[15]+=p3.w;
        }
        if (BFAST) {
            const float4* p = (const float4*)(Bp + kc + half * 8);
            __builtin_prefetch(p + 16, 0, 1);
            unpack4(bv + 0, p[0]); unpack4(bv + 4, p[1]);
            unpack4(bv + 8, p[4]); unpack4(bv + 12, p[5]);
        } else {
#pragma unroll
            for (int i = 0; i < 16; ++i) { int di = (i < 8) ? i : i + 8; bv[i] = Bv[(long)(kc + di) * bK]; }
        }
        v16h a, bb;
#pragma unroll
        for (int i = 0; i < 16; ++i) { a[i] = (_Float16)av[i]; bb[i] = (_Float16)bv[i]; }
        c = __builtin_amdgcn_wmma_f32_16x16x32_f16(false, a, false, bb, (short)0, c, false, false);
    }

#pragma unroll
    for (int r = 0; r < 8; ++r) {
        int mm = tm * 16 + r + half * 8;
        int nn = tn * 16 + nrow;
        if (mm < M && nn < N) {
            float v = c[r];
            if (biasMode == 1)      v += bias[mm];
            else if (biasMode == 2) v += bias[nn];
            Ob[(long)mm * oM + nn] = v;
        }
    }
}

// ---------------------------------------------------------------------------
// Implicit-GEMM 3x3 conv (pad=1), Cout=256, K = Cin*9 (multiple of 32)
// ---------------------------------------------------------------------------
template<int RELU>
__global__ void da_conv3x3_wmma(const float* __restrict__ X, long inBatch,
                                int Cin, int Hin, int Win,
                                const float* __restrict__ Wt, const float* __restrict__ bias,
                                float* __restrict__ Out, long oBatch, long oM,
                                int Hout, int Wout, int stride)
{
    int N = Hout * Wout;
    int K = Cin * 9;
    int lane = threadIdx.x & 31;
    int wid  = threadIdx.x >> 5;
    int half = lane >> 4;
    int nrow = lane & 15;
    int tn_count = (N + 15) >> 4;
    int tile = blockIdx.x * 4 + wid;
    if (tile >= 16 * tn_count) return;
    int tm = tile / tn_count;
    int tn = tile - tm * tn_count;

    const float* Xb = X + (long)blockIdx.y * inBatch;
    float*       Ob = Out + (long)blockIdx.y * oBatch;

    int m = tm * 16 + nrow;                       // < 256 always
    int n = tn * 16 + nrow;
    int nc = (n < N) ? n : (N - 1);
    int oh = nc / Wout;
    int ow = nc - oh * Wout;
    int ohs = oh * stride - 1;
    int ows = ow * stride - 1;
    const float* Wp = Wt + (long)m * K;

    v8f c;
#pragma unroll
    for (int j = 0; j < 8; ++j) c[j] = 0.0f;

    for (int kc = 0; kc < K; kc += 32) {
        // A: weights, contiguous
        v16h a;
        {
            const float4* p = (const float4*)(Wp + kc + half * 16);
            __builtin_prefetch(p + 16, 0, 1);
            cvt16(a, p[0], p[1], p[2], p[3]);
        }
        // B: im2col gather, branchless (incremental (ci,t) decode)
        float bv[16];
        int k0 = kc + half * 16;
        int ci = k0 / 9;
        int tt = k0 - ci * 9;
#pragma unroll
        for (int i = 0; i < 16; ++i) {
            int kh = tt / 3;
            int kw = tt - kh * 3;
            int ih = ohs + kh;
            int iw = ows + kw;
            bool vld = ((unsigned)ih < (unsigned)Hin) && ((unsigned)iw < (unsigned)Win);
            long off = ((long)ci * Hin + ih) * Win + iw;
            float x = Xb[vld ? off : 0];
            x = vld ? x : 0.0f;
            if (RELU) x = fmaxf(x, 0.0f);
            bv[i] = x;
            int wrap = (tt == 8) ? 1 : 0;
            tt = wrap ? 0 : (tt + 1);
            ci += wrap;
        }
        v16h bb;
#pragma unroll
        for (int i = 0; i < 16; ++i) bb[i] = (_Float16)bv[i];
        c = __builtin_amdgcn_wmma_f32_16x16x32_f16(false, a, false, bb, (short)0, c, false, false);
    }

#pragma unroll
    for (int r = 0; r < 8; ++r) {
        int mm = tm * 16 + r + half * 8;
        int nn = tn * 16 + nrow;
        if (nn < N)
            Ob[(long)mm * oM + nn] = c[r] + bias[mm];
    }
}

// ---------------------------------------------------------------------------
// Nearest-upsample + add (in place on dst)
// ---------------------------------------------------------------------------
__global__ void da_upsample_add(float* __restrict__ dst, const float* __restrict__ src,
                                int C, int Hd, int Wd, int Hs, int Ws)
{
    long per = (long)C * Hd * Wd;
    long idx = (long)blockIdx.x * blockDim.x + threadIdx.x;
    if (idx >= per) return;
    int w = (int)(idx % Wd);
    long t = idx / Wd;
    int h = (int)(t % Hd);
    int cch = (int)(t / Hd);
    int hs = h * Hs / Hd;
    int ws = w * Ws / Wd;
    long bd = (long)blockIdx.y * per;
    long bs = (long)blockIdx.y * C * Hs * Ws;
    dst[bd + idx] += src[bs + ((long)cch * Hs + hs) * Ws + ws];
}

// ---------------------------------------------------------------------------
// Fused cross-attention. Block = 4 waves handles one (b, 16-query tile).
//   Q tile staged in LDS (TDM tensor_load_to_lds when available),
//   scores[16,Lpad] in LDS via WMMA -> softmax (shfl) -> P@V WMMA (A from LDS),
//   mean over S shots in registers -> out channels [256,512).
// Qb: [B,HW,D] ; Kb: [B*S,L,D] ; Sv: [B*S,D,L] ; peT: [D,L]
// ---------------------------------------------------------------------------
__global__ void da_attn_wmma(const float* __restrict__ Qb,
                             const float* __restrict__ Kb,
                             const float* __restrict__ Sv,
                             const float* __restrict__ peT,
                             float* __restrict__ outp,
                             int HW, int L, int S)
{
    extern __shared__ float smem[];
    float* qT = smem;               // [16][256]
    float* sc = smem + 16 * 256;    // [16][Lpad]
    const int D = 256;
    int Lpad = (L + 31) & ~31;
    int KT16 = Lpad >> 4;
    int KT32 = Lpad >> 5;
    int KTvec = ((L & 3) == 0) ? (L >> 5) : 0;   // full-32 tiles entirely inside L
    int lane = threadIdx.x & 31;
    int wid  = threadIdx.x >> 5;
    int half = lane >> 4;
    int nrow = lane & 15;
    int q0 = blockIdx.x * 16;
    int b  = blockIdx.y;

    // ---- stage Q tile [16,256] into LDS ------------------------------------
#if DA_HAS_TDM
    if (wid == 0) {
        typedef __attribute__((ext_vector_type(4))) unsigned int u32x4;
        typedef __attribute__((ext_vector_type(8))) int i32x8;
        typedef __attribute__((ext_vector_type(4))) int i32x4;
        unsigned lds = (unsigned)(unsigned long long)(void*)qT;
        unsigned long long ga =
            (unsigned long long)(const void*)(Qb + ((long)b * HW + q0) * D);
        unsigned remRows = (unsigned)(HW - q0);      // rows beyond tensor -> zero-filled
        u32x4 g0; i32x8 g1; i32x4 gz; i32x8 gz8;
        gz[0] = gz[1] = gz[2] = gz[3] = 0;
        gz8[0] = gz8[1] = gz8[2] = gz8[3] = 0;
        gz8[4] = gz8[5] = gz8[6] = gz8[7] = 0;
        g0[0] = 1u;                                       // count=1, load descriptor
        g0[1] = lds;                                      // lds_addr
        g0[2] = (unsigned)ga;                             // global_addr[31:0]
        g0[3] = (unsigned)((ga >> 32) & 0x01FFFFFFu) | (2u << 30); // addr[56:32] | type=2
        g1[0] = 0x00020000;                               // data_size = 4B
        g1[1] = (int)(256u << 16);                        // tensor_dim0 = 256
        g1[2] = (int)((remRows & 0xFFFFu) << 16);         // tensor_dim1 (rows left)
        g1[3] = (int)(256u << 16);                        // tile_dim0 = 256
        g1[4] = 16;                                       // tile_dim1 = 16
        g1[5] = 256;                                      // tensor_dim0_stride = 256
        g1[6] = (int)(256u << 16);                        // tensor_dim1_stride = 256
        g1[7] = 0;
        __builtin_amdgcn_tensor_load_to_lds(g0, g1, gz, gz, gz8, 0);
        __builtin_amdgcn_s_wait_tensorcnt(0);
    }
#else
    for (int idx = threadIdx.x * 4; idx < 16 * 256; idx += 128 * 4) {
        int row = idx >> 8;
        int q = q0 + row;
        int qc = (q < HW) ? q : 0;
        float4 v = *(const float4*)(Qb + ((long)b * HW + qc) * D + (idx & 255));
        *(float4*)(qT + idx) = v;
    }
#endif
    __syncthreads();

    v8f macc[4];
#pragma unroll
    for (int t = 0; t < 4; ++t)
#pragma unroll
        for (int j = 0; j < 8; ++j) macc[t][j] = 0.0f;

    for (int s = 0; s < S; ++s) {
        long sb = (long)b * S + s;
        const float* Kbase = Kb + sb * (long)L * D;
        const float* Svb   = Sv + sb * (long)D * L;

        // ---- phase 1: scores = (Q K^T)/16 ----------------------------------
        for (int kt = wid; kt < KT16; kt += 4) {
            int kn = kt * 16 + nrow;
            int knc = (kn < L) ? kn : (L - 1);             // clamped, unguarded loads
            const float* Krow = Kbase + (long)knc * D;
            v8f c;
#pragma unroll
            for (int j = 0; j < 8; ++j) c[j] = 0.0f;
            for (int kc = 0; kc < D; kc += 32) {
                const float4* qp = (const float4*)(qT + nrow * D + kc + half * 16);
                const float4* kp = (const float4*)(Krow + kc + half * 16);
                v16h a, bb;
                cvt16(a, qp[0], qp[1], qp[2], qp[3]);
                cvt16(bb, kp[0], kp[1], kp[2], kp[3]);
                c = __builtin_amdgcn_wmma_f32_16x16x32_f16(false, a, false, bb, (short)0, c, false, false);
            }
#pragma unroll
            for (int r = 0; r < 8; ++r) {
                int mm = r + half * 8;
                int nn = kt * 16 + nrow;
                sc[mm * Lpad + nn] = (nn < L) ? c[r] * 0.0625f : -3.0e38f;
            }
        }
        __syncthreads();

        // ---- phase 2: row softmax ------------------------------------------
        for (int r = wid; r < 16; r += 4) {
            float* row = sc + r * Lpad;
            float mx = -3.0e38f;
            for (int j = lane; j < Lpad; j += 32) mx = fmaxf(mx, row[j]);
#pragma unroll
            for (int o = 16; o > 0; o >>= 1) mx = fmaxf(mx, __shfl_xor(mx, o, 32));
            float sum = 0.0f;
            for (int j = lane; j < Lpad; j += 32) {
                float e = __expf(row[j] - mx);
                row[j] = e;
                sum += e;
            }
#pragma unroll
            for (int o = 16; o > 0; o >>= 1) sum += __shfl_xor(sum, o, 32);
            float inv = 1.0f / sum;
            for (int j = lane; j < Lpad; j += 32) row[j] *= inv;
        }
        __syncthreads();

        // ---- phase 3: mean += (P @ (sx+pe)) / S ----------------------------
#pragma unroll
        for (int t = 0; t < 4; ++t) {
            int nt = wid + 4 * t;
            int dn = nt * 16 + nrow;
            const float* Vcol = Svb + (long)dn * L;
            const float* Pcol = peT + (long)dn * L;
            v8f c;
#pragma unroll
            for (int j = 0; j < 8; ++j) c[j] = 0.0f;
            // full vector tiles (uniform loop bound, no per-iteration tests)
            for (int kt = 0; kt < KTvec; ++kt) {
                const float4* ap = (const float4*)(sc + nrow * Lpad + kt * 32 + half * 16);
                int kbase = kt * 32 + half * 16;
                const float4* vp = (const float4*)(Vcol + kbase);
                const float4* pp = (const float4*)(Pcol + kbase);
                float4 v0 = vp[0], v1 = vp[1], v2 = vp[2], v3 = vp[3];
                float4 p0 = pp[0], p1 = pp[1], p2 = pp[2], p3 = pp[3];
                float4 s0, s1, s2, s3;
                s0.x=v0.x+p0.x; s0.y=v0.y+p0.y; s0.z=v0.z+p0.z; s0.w=v0.w+p0.w;
                s1.x=v1.x+p1.x; s1.y=v1.y+p1.y; s1.z=v1.z+p1.z; s1.w=v1.w+p1.w;
                s2.x=v2.x+p2.x; s2.y=v2.y+p2.y; s2.z=v2.z+p2.z; s2.w=v2.w+p2.w;
                s3.x=v3.x+p3.x; s3.y=v3.y+p3.y; s3.z=v3.z+p3.z; s3.w=v3.w+p3.w;
                v16h a, bb;
                cvt16(a, ap[0], ap[1], ap[2], ap[3]);
                cvt16(bb, s0, s1, s2, s3);
                c = __builtin_amdgcn_wmma_f32_16x16x32_f16(false, a, false, bb, (short)0, c, false, false);
            }
            // tail tiles (clamped scalar gather; P rows are 0 on padded k)
            for (int kt = KTvec; kt < KT32; ++kt) {
                const float4* ap = (const float4*)(sc + nrow * Lpad + kt * 32 + half * 16);
                int kbase = kt * 32 + half * 16;
                v16h a, bb;
                cvt16(a, ap[0], ap[1], ap[2], ap[3]);
#pragma unroll
                for (int i = 0; i < 16; ++i) {
                    int k = kbase + i;
                    int kcl = (k < L) ? k : (L - 1);
                    bb[i] = (_Float16)(Vcol[kcl] + Pcol[kcl]);
                }
                c = __builtin_amdgcn_wmma_f32_16x16x32_f16(false, a, false, bb, (short)0, c, false, false);
            }
            float invS = 1.0f / (float)S;
#pragma unroll
            for (int j = 0; j < 8; ++j) macc[t][j] += c[j] * invS;
        }
        __syncthreads();
    }

    // ---- store m into out channels [256..512) ------------------------------
#pragma unroll
    for (int t = 0; t < 4; ++t) {
        int nt = wid + 4 * t;
#pragma unroll
        for (int r = 0; r < 8; ++r) {
            int q = q0 + r + half * 8;
            int d = nt * 16 + nrow;
            if (q < HW)
                outp[((long)b * 512 + 256 + d) * HW + q] = macc[t][r];
        }
    }
}

// ---------------------------------------------------------------------------
// host orchestration
// ---------------------------------------------------------------------------
extern "C" void kernel_launch(void* const* d_in, const int* in_sizes, int n_in,
                              void* d_out, int out_size, void* d_ws, size_t ws_size,
                              hipStream_t stream)
{
    (void)in_sizes; (void)n_in; (void)out_size; (void)ws_size;

    const float* C3 = (const float*)d_in[0];
    const float* C4 = (const float*)d_in[1];
    const float* C5 = (const float*)d_in[2];
    const float* S3 = (const float*)d_in[3];
    const float* S4 = (const float*)d_in[4];
    const float* S5 = (const float*)d_in[5];
    const float* p3_1_w = (const float*)d_in[6];  const float* p3_1_b = (const float*)d_in[7];
    const float* p3_2_w = (const float*)d_in[8];  const float* p3_2_b = (const float*)d_in[9];
    const float* p4_1_w = (const float*)d_in[10]; const float* p4_1_b = (const float*)d_in[11];
    const float* p4_2_w = (const float*)d_in[12]; const float* p4_2_b = (const float*)d_in[13];
    const float* p5_1_w = (const float*)d_in[14]; const float* p5_1_b = (const float*)d_in[15];
    const float* p5_2_w = (const float*)d_in[16]; const float* p5_2_b = (const float*)d_in[17];
    const float* p6_w = (const float*)d_in[18];   const float* p6_b = (const float*)d_in[19];
    const float* p7_w = (const float*)d_in[20];   const float* p7_b = (const float*)d_in[21];
    const float* q_w = (const float*)d_in[22];    const float* q_b = (const float*)d_in[23];
    const float* k_w = (const float*)d_in[24];    const float* k_b = (const float*)d_in[25];

    const int B = 8, S = 5, NS = B * S;
    const int HWs[5] = {1600, 400, 100, 25, 9};

    float* out = (float*)d_out;
    long outOff[5];
    {
        long acc = 0;
        for (int l = 0; l < 5; ++l) { outOff[l] = acc; acc += (long)B * 512 * HWs[l]; }
    }

    // workspace carve-up
    float* w = (float*)d_ws;
    float* P3lat = w; w += (long)B * 256 * 1600;
    float* P4lat = w; w += (long)B * 256 * 400;
    float* P5lat = w; w += (long)B * 256 * 100;
    float* P3s = w; w += (long)NS * 256 * 1600;
    float* P4s = w; w += (long)NS * 256 * 400;
    float* P5s = w; w += (long)NS * 256 * 100;
    float* P6s = w; w += (long)NS * 256 * 25;
    float* P7s = w; w += (long)NS * 256 * 9;
    float* Qbuf[5]; for (int l = 0; l < 5; ++l) { Qbuf[l] = w; w += (long)B * HWs[l] * 256; }
    float* Kbuf[5]; for (int l = 0; l < 5; ++l) { Kbuf[l] = w; w += (long)NS * HWs[l] * 256; }
    float* peTab = w; w += 1600L * 256;
    float* peT[5]; for (int l = 0; l < 5; ++l) { peT[l] = w; w += 256L * HWs[l]; }
    float* Sbuf[5] = {P3s, P4s, P5s, P6s, P7s};

    dim3 blk(128);
    auto gemmGrid = [](int M, int N, int batches) {
        int tiles = ((M + 15) / 16) * ((N + 15) / 16);
        return dim3((unsigned)((tiles + 3) / 4), (unsigned)batches);
    };

    // 0) PE tables
    for (int l = 0; l < 5; ++l) {
        long n = (long)HWs[l] * 256;
        da_pe_gen<<<dim3((unsigned)((n + 255) / 256)), dim3(256), 0, stream>>>(
            (l == 0) ? peTab : nullptr, peT[l], HWs[l]);
    }

    // 1) lateral 1x1 convs (GEMM 256 x Cin x HW): A=weights (fast), B=activations (strided)
    da_gemm_nt_wmma<1,0,0><<<gemmGrid(256, 100, B), blk, 0, stream>>>(
        p5_1_w, 0, 2048, 1, C5, (long)2048 * 100, 1, 100, p5_1_b, 1, nullptr,
        P5lat, (long)256 * 100, 100, 256, 100, 2048);
    da_gemm_nt_wmma<1,0,0><<<gemmGrid(256, 400, B), blk, 0, stream>>>(
        p4_1_w, 0, 1024, 1, C4, (long)1024 * 400, 1, 400, p4_1_b, 1, nullptr,
        P4lat, (long)256 * 400, 400, 256, 400, 1024);
    da_gemm_nt_wmma<1,0,0><<<gemmGrid(256, 1600, B), blk, 0, stream>>>(
        p3_1_w, 0, 512, 1, C3, (long)512 * 1600, 1, 1600, p3_1_b, 1, nullptr,
        P3lat, (long)256 * 1600, 1600, 256, 1600, 512);
    da_gemm_nt_wmma<1,0,0><<<gemmGrid(256, 100, NS), blk, 0, stream>>>(
        p5_1_w, 0, 2048, 1, S5, (long)2048 * 100, 1, 100, p5_1_b, 1, nullptr,
        P5s, (long)256 * 100, 100, 256, 100, 2048);
    da_gemm_nt_wmma<1,0,0><<<gemmGrid(256, 400, NS), blk, 0, stream>>>(
        p4_1_w, 0, 1024, 1, S4, (long)1024 * 400, 1, 400, p4_1_b, 1, nullptr,
        P4s, (long)256 * 400, 400, 256, 400, 1024);
    da_gemm_nt_wmma<1,0,0><<<gemmGrid(256, 1600, NS), blk, 0, stream>>>(
        p3_1_w, 0, 512, 1, S3, (long)512 * 1600, 1, 1600, p3_1_b, 1, nullptr,
        P3s, (long)256 * 1600, 1600, 256, 1600, 512);

    // 2) P6 = conv3x3 s2 (query -> d_out pre-ReLU, support -> ws)
    da_conv3x3_wmma<0><<<gemmGrid(256, 25, B), blk, 0, stream>>>(
        C5, (long)2048 * 100, 2048, 10, 10, p6_w, p6_b,
        out + outOff[3], (long)512 * 25, 25, 5, 5, 2);
    da_conv3x3_wmma<0><<<gemmGrid(256, 25, NS), blk, 0, stream>>>(
        S5, (long)2048 * 100, 2048, 10, 10, p6_w, p6_b,
        P6s, (long)256 * 25, 25, 5, 5, 2);

    // 3) P7 = conv3x3 s2 on relu(P6)
    da_conv3x3_wmma<1><<<gemmGrid(256, 9, B), blk, 0, stream>>>(
        out + outOff[3], (long)512 * 25, 256, 5, 5, p7_w, p7_b,
        out + outOff[4], (long)512 * 9, 9, 3, 3, 2);
    da_conv3x3_wmma<1><<<gemmGrid(256, 9, NS), blk, 0, stream>>>(
        P6s, (long)256 * 25, 256, 5, 5, p7_w, p7_b,
        P7s, (long)256 * 9, 9, 3, 3, 2);

    // 4) top-down + 3x3 smoothing (writes qx halves of d_out)
    da_upsample_add<<<dim3(400, B), dim3(256), 0, stream>>>(P4lat, P5lat, 256, 20, 20, 10, 10);
    da_conv3x3_wmma<0><<<gemmGrid(256, 100, B), blk, 0, stream>>>(
        P5lat, (long)256 * 100, 256, 10, 10, p5_2_w, p5_2_b,
        out + outOff[2], (long)512 * 100, 100, 10, 10, 1);
    da_upsample_add<<<dim3(1600, B), dim3(256), 0, stream>>>(P3lat, P4lat, 256, 40, 40, 20, 20);
    da_conv3x3_wmma<0><<<gemmGrid(256, 400, B), blk, 0, stream>>>(
        P4lat, (long)256 * 400, 256, 20, 20, p4_2_w, p4_2_b,
        out + outOff[1], (long)512 * 400, 400, 20, 20, 1);
    da_conv3x3_wmma<0><<<gemmGrid(256, 1600, B), blk, 0, stream>>>(
        P3lat, (long)256 * 1600, 256, 40, 40, p3_2_w, p3_2_b,
        out + outOff[0], (long)512 * 1600, 1600, 40, 40, 1);

    // 5) Q projections (qx from d_out) and K projections (+PE table)
    for (int l = 0; l < 5; ++l) {
        int HW = HWs[l];
        da_gemm_nt_wmma<0,1,0><<<gemmGrid(HW, 256, B), blk, 0, stream>>>(
            out + outOff[l], (long)512 * HW, 1, HW,
            q_w, 0, 256, 1, q_b, 2, nullptr,
            Qbuf[l], (long)HW * 256, 256, HW, 256, 256);
        da_gemm_nt_wmma<0,1,1><<<gemmGrid(HW, 256, NS), blk, 0, stream>>>(
            Sbuf[l], (long)256 * HW, 1, HW,
            k_w, 0, 256, 1, k_b, 2, peTab,
            Kbuf[l], (long)HW * 256, 256, HW, 256, 256);
    }

    // 6) fused attention -> m halves of d_out
    for (int l = 0; l < 5; ++l) {
        int HW = HWs[l];
        int Lpad = (HW + 31) & ~31;
        size_t sh = (size_t)(16 * 256 + 16 * Lpad) * sizeof(float);
        dim3 g((unsigned)((HW + 15) / 16), (unsigned)B);
        da_attn_wmma<<<g, blk, sh, stream>>>(Qbuf[l], Kbuf[l], Sbuf[l], peT[l],
                                             out + outOff[l], HW, HW, S);
    }
}
